// CAGKE_1_54348516164260
// MI455X (gfx1250) — matrix operand order repaired and checked by reading, not compiled, not run on hardware
//
#include <hip/hip_runtime.h>
#include <hip/hip_bf16.h>

typedef __attribute__((ext_vector_type(16))) _Float16 v16h;
typedef __attribute__((ext_vector_type(8)))  float    v8f;

#define T_N     8192
#define E_N     128
#define KTAPS   160      // 129 live taps padded to 5 chunks of 32
#define LG      80       // left zero-guard: mp[i] = mask[i - LG]
#define MPLEN   8448     // guarded mask length in halves (max read idx 8365)
#define MP_OFF  512      // byte offset of mp inside d_ws (kf occupies [0,320))
#define INV_SQRT_2PI 0.39894228f

// ---------------------------------------------------------------------------
// Kernel 1: fold softmax(weight) over E sigmas into one 160-tap f16 kernel.
//   Kf[u] = K(64 - u),  K(d) = sum_e w_e * (1/(sigma_e*sqrt(2pi))) * exp(-d^2/(2 sigma_e^2))
// out[t] = sum_u Kf[u] * mask[t + u - 65] + noise[t]
// ---------------------------------------------------------------------------
__global__ void CAGKE_prep_kernel(const float* __restrict__ sigma,
                                  const float* __restrict__ weight,
                                  _Float16* __restrict__ kf)
{
    __shared__ float w[E_N];
    __shared__ float sg[E_N];
    const int tid = threadIdx.x;
    if (tid < E_N) { w[tid] = weight[tid]; sg[tid] = sigma[tid]; }
    __syncthreads();
    if (tid == 0) {   // E=128 is tiny; serial softmax is noise-level cost
        float mx = w[0];
        for (int e = 1; e < E_N; ++e) mx = fmaxf(mx, w[e]);
        float s = 0.0f;
        for (int e = 0; e < E_N; ++e) { w[e] = __expf(w[e] - mx); s += w[e]; }
        const float inv = 1.0f / s;
        for (int e = 0; e < E_N; ++e) w[e] *= inv;
    }
    __syncthreads();
    for (int u = tid; u < KTAPS; u += blockDim.x) {
        const float d  = 64.0f - (float)u;
        const float d2 = d * d;
        float acc = 0.0f;
        for (int e = 0; e < E_N; ++e) {
            const float s = sg[e];
            acc += w[e] * (INV_SQRT_2PI / s) * __expf(-d2 / (2.0f * s * s));
        }
        kf[u] = (_Float16)acc;   // taps < 6e-8 flush to 0 in f16: < 1e-5 abs error
    }
}

// ---------------------------------------------------------------------------
// Kernel 2: zero-guarded f16 mask  mp[i] = (i in [LG, LG+T)) ? (X[i-LG] > 0.5) : 0
// ---------------------------------------------------------------------------
__global__ void CAGKE_mask_kernel(const float* __restrict__ X,
                                  _Float16* __restrict__ mp)
{
    const int i = blockIdx.x * blockDim.x + threadIdx.x;
    if (i < MPLEN) {
        float v = 0.0f;
        const int t = i - LG;
        if (t >= 0 && t < T_N) v = (X[t] > 0.5f) ? 1.0f : 0.0f;
        mp[i] = (_Float16)v;
    }
}

// ---------------------------------------------------------------------------
// Kernel 3: im2col conv via v_wmma_f32_16x16x32_f16.
//   Each wave32 computes one 16-output tile; 5 WMMAs reduce 160 taps.
//   A (16x32): kernel taps replicated across rows (contiguous k = e + 16*hi, same order as B).
//   B (32x16): B[k,n] = mp[t0+n + c*32 + k + 15]; lane element e -> k = e + 16*hi
//              => contiguous 16-half run per lane (2B-aligned element loads).
//   D rows are identical; lanes with hi==0 hold out[t0 + (lane&15)] in acc[0].
// ---------------------------------------------------------------------------
__global__ void __launch_bounds__(256)
CAGKE_conv_kernel(const _Float16* __restrict__ kf,
                  const _Float16* __restrict__ mp,
                  const float* __restrict__ noise,
                  float* __restrict__ out)
{
    const int lane = threadIdx.x & 31;
    const int wave = threadIdx.x >> 5;
    const int tile = blockIdx.x * 8 + wave;   // 64 blocks * 8 waves = 512 tiles
    const int t0   = tile * 16;
    const int n    = lane & 15;
    const int hi   = lane >> 4;

    v8f acc = {0.f, 0.f, 0.f, 0.f, 0.f, 0.f, 0.f, 0.f};

#pragma unroll
    for (int c = 0; c < 5; ++c) {
        v16h a, b;
        // A fragment: halves 0..7 <- Kf[c*32 + 16*hi + e]; 8..15 <- Kf[c*32+16*hi+8 + e]
        const _Float16* alo = kf + c * 32 + 16 * hi;
        const _Float16* ahi = kf + c * 32 + 16 * hi + 8;
#pragma unroll
        for (int e = 0; e < 8; ++e) { a[e] = alo[e]; a[e + 8] = ahi[e]; }
        // B fragment: contiguous run in the guarded mask
        const _Float16* bp = mp + t0 + n + c * 32 + 16 * hi + 15;
#pragma unroll
        for (int e = 0; e < 16; ++e) b[e] = bp[e];

        // 8 args: (neg_a, A, neg_b, B, c_mod, C, reuse_a, reuse_b)
        acc = __builtin_amdgcn_wmma_f32_16x16x32_f16(
            false, a, false, b, (short)0, acc, false, false);
    }

    if (hi == 0) {
        const int t = t0 + n;
        out[t] = acc[0] + noise[t];
    }
}

// ---------------------------------------------------------------------------
// inputs (setup_inputs order): X[8192] f32, sigma[128] f32, weight[128] f32,
// noise[8192] f32.  output: 8192 f32.  ws usage: ~17.4 KB.
// ---------------------------------------------------------------------------
extern "C" void kernel_launch(void* const* d_in, const int* in_sizes, int n_in,
                              void* d_out, int out_size, void* d_ws, size_t ws_size,
                              hipStream_t stream)
{
    const float* X      = (const float*)d_in[0];
    const float* sigma  = (const float*)d_in[1];
    const float* weight = (const float*)d_in[2];
    const float* noise  = (const float*)d_in[3];
    float*       out    = (float*)d_out;

    _Float16* kf = (_Float16*)d_ws;                       // 160 halves
    _Float16* mp = (_Float16*)((char*)d_ws + MP_OFF);     // 8448 halves

    CAGKE_prep_kernel<<<1, 256, 0, stream>>>(sigma, weight, kf);
    CAGKE_mask_kernel<<<(MPLEN + 255) / 256, 256, 0, stream>>>(X, mp);
    CAGKE_conv_kernel<<<(T_N / 16) / 8, 256, 0, stream>>>(kf, mp, noise, out);
}
